// GATLayer_19499151524590
// MI455X (gfx1250) — matrix operand order repaired and checked by reading, not compiled, not run on hardware
//
#include <hip/hip_runtime.h>

#define NN 100000
#define NE 1600000
#define NEG_SLOPE 0.2f

typedef __attribute__((ext_vector_type(16))) __bf16   v16bf;
typedef __attribute__((ext_vector_type(8)))  float    v8f;
typedef __attribute__((ext_vector_type(4)))  float    f4;
typedef __attribute__((ext_vector_type(8)))  unsigned v8u;
typedef __attribute__((ext_vector_type(4)))  unsigned u32x4;
typedef __attribute__((ext_vector_type(8)))  int      i32x8;
typedef __attribute__((ext_vector_type(4)))  int      i32x4;

#if defined(__has_builtin)
#if __has_builtin(__builtin_amdgcn_tensor_load_to_lds)
#define HAVE_TDM 1
#endif
#endif

// ---------------- helpers ----------------

// round-to-nearest-even f32->bf16 for a pair, packed into one dword
__device__ __forceinline__ unsigned pk2(float f0, float f1) {
    unsigned u0 = __float_as_uint(f0), u1 = __float_as_uint(f1);
    u0 += 0x7fffu + ((u0 >> 16) & 1u);
    u1 += 0x7fffu + ((u1 >> 16) & 1u);
    return __builtin_amdgcn_perm(u1, u0, 0x07060302); // {u1.hi16, u0.hi16}
}

__device__ __forceinline__ v16bf pack16(f4 q0, f4 q1, f4 q2, f4 q3) {
    v8u p;
    p[0] = pk2(q0.x, q0.y); p[1] = pk2(q0.z, q0.w);
    p[2] = pk2(q1.x, q1.y); p[3] = pk2(q1.z, q1.w);
    p[4] = pk2(q2.x, q2.y); p[5] = pk2(q2.z, q2.w);
    p[6] = pk2(q3.x, q3.y); p[7] = pk2(q3.z, q3.w);
    return __builtin_bit_cast(v16bf, p);
}

__device__ __forceinline__ v8f wmma_bf16(v16bf a, v16bf b, v8f c) {
    return __builtin_amdgcn_wmma_f32_16x16x32_bf16(
        false, a, false, b, (short)0, c, false, false);
}

// A tile rows [row0,row0+16), K slice [k0,k0+32): lanes 0-15 row=lane,
// K runs {kh..kh+7, 16+kh..23+kh} with kh=0; lanes 16-31 kh=8. Two
// contiguous 8-float runs -> 4x b128 loads.
__device__ __forceinline__ v16bf load_a16x32v(const float* A, int lda,
                                              long row0, int k0, int lane) {
    long row = row0 + (lane & 15);
    int kh = (lane >> 4) << 3;
    const float* p = A + (size_t)row * lda + k0 + kh;
    f4 q0 = *(const f4*)(p);
    f4 q1 = *(const f4*)(p + 4);
    f4 q2 = *(const f4*)(p + 16);
    f4 q3 = *(const f4*)(p + 20);
    return pack16(q0, q1, q2, q3);
}

// pre-packed B tile fetch: tile index (kt*2+ct), 32B per lane (2x b128)
__device__ __forceinline__ v16bf bload(const v8u* bp, int kt, int ct, int lane) {
    return __builtin_bit_cast(v16bf, bp[(size_t)((kt * 2 + ct) * 32 + lane)]);
}

__device__ __forceinline__ void store_d16x16(float* D, int ldd, long row0,
                                             int col0, int lane, v8f d) {
    int col = col0 + (lane & 15);
    int rh  = (lane >> 4) << 3;
#pragma unroll
    for (int j = 0; j < 8; ++j)
        D[(size_t)(row0 + rh + j) * ldd + col] = d[j];
}

__device__ __forceinline__ void atomicMaxF(float* addr, float val) {
    int* ia = (int*)addr;
    int old = *ia;
    while (__int_as_float(old) < val) {
        int prev = atomicCAS(ia, old, __float_as_int(val));
        if (prev == old) break;
        old = prev;
    }
}

// ---------------- kernels ----------------

__global__ void k_fill(float* p, float v, long n) {
    long i = (long)blockIdx.x * blockDim.x + threadIdx.x;
    if (i < n) p[i] = v;
}

// Pack weight W[K,32] into per-lane WMMA B images, nk K-tiles x 2 col-tiles.
// Element i of lane image = B[k0 + ((lane>>4)<<3) + i + (i>=8 ? 8 : 0)][col0 + (lane&15)].
__global__ void k_pack_b(const float* __restrict__ W, int nk,
                         v8u* __restrict__ outp) {
    int t = blockIdx.x * blockDim.x + threadIdx.x;
    int total = nk * 2 * 32;
    if (t >= total) return;
    int lane = t & 31;
    int tile = t >> 5;
    int ct = tile & 1, kt = tile >> 1;
    int col = ct * 16 + (lane & 15);
    int k0  = kt * 32 + ((lane >> 4) << 3);
    v8u p;
#pragma unroll
    for (int i = 0; i < 8; ++i)
        p[i] = pk2(W[(size_t)(k0 + 2 * i + (i >= 4 ? 8 : 0)) * 32 + col],
                   W[(size_t)(k0 + 2 * i + 1 + (i >= 4 ? 8 : 0)) * 32 + col]);
    outp[t] = p;
}

// h = x@W_lin, res = x@W_res   (x: [NN,64])
__global__ void k_node_gemm(const float* __restrict__ x,
                            const v8u* __restrict__ bp_lin,
                            const v8u* __restrict__ bp_res,
                            float* __restrict__ h, float* __restrict__ res,
                            int ntiles) {
    int tile = blockIdx.x * (blockDim.x >> 5) + (threadIdx.x >> 5);
    if (tile >= ntiles) return;
    int lane = threadIdx.x & 31;
    long row0 = (long)tile * 16;
    v16bf a0 = load_a16x32v(x, 64, row0, 0,  lane);
    v16bf a1 = load_a16x32v(x, 64, row0, 32, lane);
#pragma unroll
    for (int ct = 0; ct < 2; ++ct) {
        v8f c = {};
        c = wmma_bf16(a0, bload(bp_lin, 0, ct, lane), c);
        c = wmma_bf16(a1, bload(bp_lin, 1, ct, lane), c);
        store_d16x16(h, 32, row0, ct * 16, lane, c);
        v8f c2 = {};
        c2 = wmma_bf16(a0, bload(bp_res, 0, ct, lane), c2);
        c2 = wmma_bf16(a1, bload(bp_res, 1, ct, lane), c2);
        store_d16x16(res, 32, row0, ct * 16, lane, c2);
    }
}

// e = edge_attr@W_lin_edge  -- A tile staged in LDS via Tensor Data Mover
__global__ void k_edge_gemm(const float* __restrict__ ea,
                            const v8u* __restrict__ bp,
                            float* __restrict__ evec, int ntiles) {
    int wv = threadIdx.x >> 5;
    int lane = threadIdx.x & 31;
    int tile = blockIdx.x * (blockDim.x >> 5) + wv;
    if (tile >= ntiles) return;
    long row0 = (long)tile * 16;
    v16bf a;
#if HAVE_TDM
    __shared__ float sh[4][512];  // one 16x32 f32 tile per wave
    {
        unsigned lds = (unsigned)(unsigned long long)(uintptr_t)&sh[wv][0];
        unsigned long long ga =
            (unsigned long long)(uintptr_t)(ea + (size_t)row0 * 32);
        u32x4 g0 = {};
        g0[0] = 1u;                                   // count=1 (user desc)
        g0[1] = lds;                                  // lds_addr
        g0[2] = (unsigned)ga;                         // global_addr[31:0]
        g0[3] = ((unsigned)(ga >> 32) & 0x01FFFFFFu)  // global_addr[56:32]
              | (2u << 30);                           // type=2 (image)
        i32x8 g1 = {};
        g1[0] = 2 << 16;                 // data_size = 4B
        g1[1] = (int)(512u << 16);       // tensor_dim0[15:0] = 512 (bits 63:48)
        g1[3] = (int)(512u << 16);       // tile_dim0 = 512     (bits 127:112)
        i32x4 z4 = {};
        i32x8 z8 = {};
        __builtin_amdgcn_tensor_load_to_lds(g0, g1, z4, z4, z8, 0);
        __builtin_amdgcn_s_wait_tensorcnt(0);
    }
    {
        int row = lane & 15;
        int kh  = (lane >> 4) << 3;
        const float* p = &sh[wv][row * 32 + kh];
        f4 q0 = *(const f4*)(p);
        f4 q1 = *(const f4*)(p + 4);
        f4 q2 = *(const f4*)(p + 16);
        f4 q3 = *(const f4*)(p + 20);
        a = pack16(q0, q1, q2, q3);
    }
#else
    a = load_a16x32v(ea, 32, row0, 0, lane);
#endif
#pragma unroll
    for (int ct = 0; ct < 2; ++ct) {
        v8f c = {};
        c = wmma_bf16(a, bload(bp, 0, ct, lane), c);
        store_d16x16(evec, 32, row0, ct * 16, lane, c);
    }
}

// alpha_raw = leaky_relu([h_i|h_j|e] @ W_node_attn); atomic max into m1[dst]
__global__ void k_alpha_logit(const long long* __restrict__ ei,
                              const float* __restrict__ h,
                              const float* __restrict__ evec,
                              const float* __restrict__ Wa,
                              float* __restrict__ alpha,
                              float* __restrict__ m1) {
    long e = (long)blockIdx.x * blockDim.x + threadIdx.x;
    if (e >= NE) return;
    long s = ei[e], d = ei[NE + e];
    const float* hi = h + d * 32;
    const float* hj = h + s * 32;
    const float* ee = evec + e * 32;
    float acc = 0.f;
#pragma unroll
    for (int f = 0; f < 32; ++f)
        acc += hi[f] * Wa[f] + hj[f] * Wa[32 + f] + ee[f] * Wa[64 + f];
    acc = acc > 0.f ? acc : NEG_SLOPE * acc;
    alpha[e] = acc;
    atomicMaxF(&m1[d], acc);
}

__global__ void k_expsum(const long long* __restrict__ ei,
                         float* __restrict__ val,
                         const float* __restrict__ m,
                         float* __restrict__ s) {
    long e = (long)blockIdx.x * blockDim.x + threadIdx.x;
    if (e >= NE) return;
    long d = ei[NE + e];
    float ex = __expf(val[e] - m[d]);
    val[e] = ex;
    atomicAdd(&s[d], ex);
}

// alpha normalize; aggr[dst] += alpha * cat(h_j, e)
__global__ void k_scatter(const long long* __restrict__ ei,
                          const float* __restrict__ h,
                          const float* __restrict__ evec,
                          float* __restrict__ alpha,
                          const float* __restrict__ s1,
                          float* __restrict__ aggr) {
    long e = (long)blockIdx.x * blockDim.x + threadIdx.x;
    if (e >= NE) return;
    long s = ei[e], d = ei[NE + e];
    float a = alpha[e] / (s1[d] + 1e-16f);
    alpha[e] = a;
    const float* hj = h + s * 32;
    const float* ee = evec + e * 32;
    float* ag = aggr + d * 64;
#pragma unroll
    for (int f = 0; f < 32; ++f) {
        atomicAdd(&ag[f],      a * hj[f]);
        atomicAdd(&ag[32 + f], a * ee[f]);
    }
}

// node_out = aggr@W_node_update; x_out = node_out + res + bias
__global__ void k_node_update(const float* __restrict__ aggr,
                              const v8u* __restrict__ bp,
                              const float* __restrict__ res,
                              const float* __restrict__ bias,
                              float* __restrict__ node_out,
                              float* __restrict__ x_out, int ntiles) {
    int tile = blockIdx.x * (blockDim.x >> 5) + (threadIdx.x >> 5);
    if (tile >= ntiles) return;
    int lane = threadIdx.x & 31;
    long row0 = (long)tile * 16;
    v16bf a0 = load_a16x32v(aggr, 64, row0, 0,  lane);
    v16bf a1 = load_a16x32v(aggr, 64, row0, 32, lane);
#pragma unroll
    for (int ct = 0; ct < 2; ++ct) {
        v8f c = {};
        c = wmma_bf16(a0, bload(bp, 0, ct, lane), c);
        c = wmma_bf16(a1, bload(bp, 1, ct, lane), c);
        int col = ct * 16 + (lane & 15);
        int rh  = (lane >> 4) << 3;
        float b = bias[col];
#pragma unroll
        for (int j = 0; j < 8; ++j) {
            size_t idx = (size_t)(row0 + rh + j) * 32 + col;
            node_out[idx] = c[j];
            x_out[idx]    = c[j] + res[idx] + b;
        }
    }
}

// out = node_out@W_out
__global__ void k_out_gemm(const float* __restrict__ node_out,
                           const v8u* __restrict__ bp,
                           float* __restrict__ out, int ntiles) {
    int tile = blockIdx.x * (blockDim.x >> 5) + (threadIdx.x >> 5);
    if (tile >= ntiles) return;
    int lane = threadIdx.x & 31;
    long row0 = (long)tile * 16;
    v16bf a = load_a16x32v(node_out, 32, row0, 0, lane);
#pragma unroll
    for (int ct = 0; ct < 2; ++ct) {
        v8f c = {};
        c = wmma_bf16(a, bload(bp, 0, ct, lane), c);
        store_d16x16(out, 32, row0, ct * 16, lane, c);
    }
}

// beta_raw = leaky_relu([out_i|out_j] @ W_edge_attn); atomic max into m2[dst]
__global__ void k_beta_logit(const long long* __restrict__ ei,
                             const float* __restrict__ out,
                             const float* __restrict__ Wb,
                             float* __restrict__ beta,
                             float* __restrict__ m2) {
    long e = (long)blockIdx.x * blockDim.x + threadIdx.x;
    if (e >= NE) return;
    long s = ei[e], d = ei[NE + e];
    const float* oi = out + d * 32;
    const float* oj = out + s * 32;
    float acc = 0.f;
#pragma unroll
    for (int f = 0; f < 32; ++f)
        acc += oi[f] * Wb[f] + oj[f] * Wb[32 + f];
    acc = acc > 0.f ? acc : NEG_SLOPE * acc;
    beta[e] = acc;
    atomicMaxF(&m2[d], acc);
}

// edge_final = (beta*alpha*cat(h_j,e))@W_edge_update + edge_attr@W_res_edge + bias
__global__ void k_edge_out(const long long* __restrict__ ei,
                           const float* __restrict__ ea,
                           const float* __restrict__ evec,
                           const float* __restrict__ h,
                           const float* __restrict__ alpha,
                           const float* __restrict__ beta,
                           const float* __restrict__ s2,
                           const v8u* __restrict__ bp_eu,
                           const v8u* __restrict__ bp_re,
                           const float* __restrict__ bias,
                           float* __restrict__ edge_final, int ntiles) {
    int tile = blockIdx.x * (blockDim.x >> 5) + (threadIdx.x >> 5);
    if (tile >= ntiles) return;
    int lane = threadIdx.x & 31;
    long row0 = (long)tile * 16;
    long row  = row0 + (lane & 15);
    long src  = ei[row];
    long dst  = ei[NE + row];
    float cf = alpha[row] * (beta[row] / (s2[dst] + 1e-16f));

    int kh = (lane >> 4) << 3;
    const float* hj = h    + src * 32 + kh;
    const float* ee = evec + row * 32 + kh;
    const float* er = ea   + row * 32 + kh;
    v16bf a0 = pack16(cf * *(const f4*)(hj),      cf * *(const f4*)(hj + 4),
                      cf * *(const f4*)(hj + 16), cf * *(const f4*)(hj + 20));
    v16bf a1 = pack16(cf * *(const f4*)(ee),      cf * *(const f4*)(ee + 4),
                      cf * *(const f4*)(ee + 16), cf * *(const f4*)(ee + 20));
    v16bf a2 = pack16(*(const f4*)(er),      *(const f4*)(er + 4),
                      *(const f4*)(er + 16), *(const f4*)(er + 20));
#pragma unroll
    for (int ct = 0; ct < 2; ++ct) {
        v8f acc = {};
        acc = wmma_bf16(a0, bload(bp_eu, 0, ct, lane), acc);
        acc = wmma_bf16(a1, bload(bp_eu, 1, ct, lane), acc);
        acc = wmma_bf16(a2, bload(bp_re, 0, ct, lane), acc);
        int col = ct * 16 + (lane & 15);
        int rh  = (lane >> 4) << 3;
        float b = bias[col];
#pragma unroll
        for (int j = 0; j < 8; ++j)
            edge_final[(size_t)(row0 + rh + j) * 32 + col] = acc[j] + b;
    }
}

// ---------------- launch ----------------

extern "C" void kernel_launch(void* const* d_in, const int* in_sizes, int n_in,
                              void* d_out, int out_size, void* d_ws, size_t ws_size,
                              hipStream_t stream) {
    (void)in_sizes; (void)n_in; (void)out_size; (void)ws_size;
    const float*     x       = (const float*)d_in[0];
    const long long* ei      = (const long long*)d_in[1];
    const float*     ea      = (const float*)d_in[2];
    const float*     W_lin   = (const float*)d_in[3];
    const float*     W_line  = (const float*)d_in[4];
    const float*     W_nattn = (const float*)d_in[5];
    const float*     W_nupd  = (const float*)d_in[6];
    const float*     W_out   = (const float*)d_in[7];
    const float*     W_eattn = (const float*)d_in[8];
    const float*     W_eupd  = (const float*)d_in[9];
    const float*     W_res   = (const float*)d_in[10];
    const float*     W_rese  = (const float*)d_in[11];
    const float*     b_node  = (const float*)d_in[12];
    const float*     b_edge  = (const float*)d_in[13];

    float* x_out = (float*)d_out;             // [NN,32]
    float* e_out = x_out + (size_t)NN * 32;   // [NE,32]

    // workspace layout (floats)
    float* ws       = (float*)d_ws;
    float* h        = ws;                           // NN*32
    float* res      = h        + (size_t)NN * 32;
    float* node_out = res      + (size_t)NN * 32;
    float* out      = node_out + (size_t)NN * 32;
    float* aggr     = out      + (size_t)NN * 32;   // NN*64
    float* m1       = aggr     + (size_t)NN * 64;
    float* s1       = m1 + NN;
    float* m2       = s1 + NN;
    float* s2       = m2 + NN;
    float* alpha    = s2 + NN;                      // NE
    float* beta     = alpha + NE;                   // NE
    float* evec     = beta  + NE;                   // NE*32
    // pre-packed B tiles (v8u = 32B per lane-image)
    v8u* bp_lin  = (v8u*)(evec + (size_t)NE * 32);  // 4 tiles = 128
    v8u* bp_res  = bp_lin  + 128;                   // 4 tiles
    v8u* bp_line = bp_res  + 128;                   // 2 tiles = 64
    v8u* bp_nupd = bp_line + 64;                    // 4 tiles
    v8u* bp_out  = bp_nupd + 128;                   // 2 tiles
    v8u* bp_eu   = bp_out  + 64;                    // 4 tiles
    v8u* bp_re   = bp_eu   + 128;                   // 2 tiles

    const int THR = 256;
    const int node_tiles = NN / 16;                 // 6250
    const int edge_tiles = NE / 16;                 // 100000
    const int WPB = 4;                              // waves per block
    dim3 gblk(128);
    dim3 gnode((node_tiles + WPB - 1) / WPB);
    dim3 gedge((edge_tiles + WPB - 1) / WPB);
    dim3 epb((NE + THR - 1) / THR);

    // pack weights into WMMA B-lane images
    k_pack_b<<<dim3(2), 128, 0, stream>>>(W_lin,  2, bp_lin);
    k_pack_b<<<dim3(2), 128, 0, stream>>>(W_res,  2, bp_res);
    k_pack_b<<<dim3(1), 128, 0, stream>>>(W_line, 1, bp_line);
    k_pack_b<<<dim3(2), 128, 0, stream>>>(W_nupd, 2, bp_nupd);
    k_pack_b<<<dim3(1), 128, 0, stream>>>(W_out,  1, bp_out);
    k_pack_b<<<dim3(2), 128, 0, stream>>>(W_eupd, 2, bp_eu);
    k_pack_b<<<dim3(1), 128, 0, stream>>>(W_rese, 1, bp_re);

    // init reductions
    k_fill<<<dim3(((long)NN * 64 + THR - 1) / THR), THR, 0, stream>>>(aggr, 0.f, (long)NN * 64);
    k_fill<<<dim3((NN + THR - 1) / THR), THR, 0, stream>>>(m1, -3.0e38f, NN);
    k_fill<<<dim3((NN + THR - 1) / THR), THR, 0, stream>>>(s1, 0.f, NN);
    k_fill<<<dim3((NN + THR - 1) / THR), THR, 0, stream>>>(m2, -3.0e38f, NN);
    k_fill<<<dim3((NN + THR - 1) / THR), THR, 0, stream>>>(s2, 0.f, NN);

    // dense projections (WMMA)
    k_node_gemm<<<gnode, gblk, 0, stream>>>(x, bp_lin, bp_res, h, res, node_tiles);
    k_edge_gemm<<<gedge, gblk, 0, stream>>>(ea, bp_line, evec, edge_tiles);

    // node attention softmax + scatter
    k_alpha_logit<<<epb, THR, 0, stream>>>(ei, h, evec, W_nattn, alpha, m1);
    k_expsum<<<epb, THR, 0, stream>>>(ei, alpha, m1, s1);
    k_scatter<<<epb, THR, 0, stream>>>(ei, h, evec, alpha, s1, aggr);

    // node update (WMMA) + fused residual output
    k_node_update<<<gnode, gblk, 0, stream>>>(aggr, bp_nupd, res, b_node, node_out, x_out, node_tiles);
    k_out_gemm<<<gnode, gblk, 0, stream>>>(node_out, bp_out, out, node_tiles);

    // edge attention softmax
    k_beta_logit<<<epb, THR, 0, stream>>>(ei, out, W_eattn, beta, m2);
    k_expsum<<<epb, THR, 0, stream>>>(ei, beta, m2, s2);

    // edge output (WMMA, fused residual GEMM + bias)
    k_edge_out<<<gedge, gblk, 0, stream>>>(ei, ea, evec, h, alpha, beta, s2,
                                           bp_eu, bp_re, b_edge, e_out, edge_tiles);
}
